// GCN_9096740733375
// MI455X (gfx1250) — matrix operand order repaired and checked by reading, not compiled, not run on hardware
//
#include <hip/hip_runtime.h>
#include <hip/hip_bf16.h>
#include <math.h>
#include <stdint.h>

// ---------------------------------------------------------------------------
// GCN inference for CDNA5 / gfx1250 (wave32, WMMA bf16 16x16x32)
// All node-feature buffers use stride S=32 floats (128B-aligned rows) and are
// zero-padded beyond the live feature count, so the WMMA K-loops for layers
// 2/3 (and the bulk of layer 1) run with no per-element guards or branches.
// Streaming inputs (x, edge_index) use non-temporal loads to keep the 192MB
// L2 reserved for the reused 12.8MB feature buffers + packed weights.
// ---------------------------------------------------------------------------

typedef __attribute__((ext_vector_type(16))) __bf16 v16bf;
typedef __attribute__((ext_vector_type(8)))  float  v8f;
typedef __attribute__((ext_vector_type(4)))  float  v4f;

#define STRIDE 32

__device__ __forceinline__ float atomic_add_f32(float* p, float v) {
    // relaxed + agent scope -> hardware global_atomic_add_f32 (resolved in L2)
    return __hip_atomic_fetch_add(p, v, __ATOMIC_RELAXED, __HIP_MEMORY_SCOPE_AGENT);
}

// ---------------- degree / normalization -----------------------------------

__global__ void k_init_deg(float* __restrict__ deg, int N) {
    int i = blockIdx.x * blockDim.x + threadIdx.x;
    if (i < N) deg[i] = 1.0f;  // self loop contributes 1
}

__global__ void k_accum_deg(float* __restrict__ deg,
                            const long long* __restrict__ ei, long long E) {
    long long e = (long long)blockIdx.x * blockDim.x + threadIdx.x;
    if (e < E) {
        int d = (int)__builtin_nontemporal_load(&ei[E + e]);
        atomic_add_f32(&deg[d], 1.0f);
    }
}

__global__ void k_rsqrt(float* __restrict__ deg, int N) {
    int i = blockIdx.x * blockDim.x + threadIdx.x;
    if (i < N) deg[i] = rsqrtf(deg[i]);
}

// ---------------- W pre-pack into B-fragment layout -------------------------
// Wp[((ct*ksteps + ks)*32 + lane)*16 + j] = bf16(W[kk*M + n]) or 0
//   n  = ct*16 + (lane&15)
//   kk = ks*32 + (j<8 ? (lane>>4)*8 + j : 16 + (lane>>4)*8 + (j-8))  (matches A fragment K order)
__global__ void k_pack_w(__bf16* __restrict__ Wp, const float* __restrict__ W,
                         int K, int M, int ksteps, int nCT) {
    int idx = blockIdx.x * blockDim.x + threadIdx.x;
    int total = nCT * ksteps * 32 * 16;
    if (idx >= total) return;
    int j    = idx & 15;
    int lane = (idx >> 4) & 31;
    int ks   = (idx >> 9) % ksteps;
    int ct   = (idx >> 9) / ksteps;
    int n  = ct * 16 + (lane & 15);
    int kk = ks * 32 + ((j < 8) ? ((lane >> 4) * 8 + j) : (16 + (lane >> 4) * 8 + (j - 8)));
    float v = (kk < K && n < M) ? W[(long long)kk * M + n] : 0.0f;
    Wp[idx] = (__bf16)v;
}

// ---------------- WMMA GEMM: Out[N x M] = A[N x K] * W[K x M] ---------------
// One wave per 16-row tile, NCT 16-wide column tiles per wave (A fragment is
// loaded once, feeds NCT WMMAs). nFull K-steps are branch-free; at most one
// guarded tail step (layer 1: K=1433 -> 44 full + 1 tail).
// A fragment (16-bit 16x32): lane m=lane&15 is row; a[0..7] = K half*8+0..7,
// a[8..15] = K 16+half*8+0..7 (half = lane>>4) -> two contiguous 8-float runs.
template <int NCT, bool NTA>
__global__ void k_gemm_wmma(const float* __restrict__ A, int lda,
                            const __bf16* __restrict__ Wp,
                            int K, int nFull, int ksteps, int M,
                            float* __restrict__ Out, int ldo,
                            int N, int nRowTiles) {
    int wave = threadIdx.x >> 5;
    int rt = blockIdx.x * (blockDim.x >> 5) + wave;
    if (rt >= nRowTiles) return;                 // wave-uniform exit
    int lane = threadIdx.x & 31;
    int m    = lane & 15;
    int half = lane >> 4;
    int row  = rt * 16 + m;
    const float* Arow = A + (long long)(row < N ? row : 0) * lda;

    v8f acc[NCT];
    #pragma unroll
    for (int c = 0; c < NCT; ++c) acc[c] = (v8f){};

    // ---- branch-free main loop ----
    for (int ks = 0; ks < nFull; ++ks) {
        const v4f* p0 = (const v4f*)(Arow + ks * 32 + half * 8);
        v4f f0, f1, f2, f3;
        if (NTA) {   // streaming input: non-temporal, don't pollute L2
            f0 = __builtin_nontemporal_load(p0 + 0);
            f1 = __builtin_nontemporal_load(p0 + 1);
            f2 = __builtin_nontemporal_load(p0 + 4);
            f3 = __builtin_nontemporal_load(p0 + 5);
        } else {     // L2-resident padded buffer
            f0 = p0[0]; f1 = p0[1]; f2 = p0[4]; f3 = p0[5];
        }
        v16bf a;
        #pragma unroll
        for (int t = 0; t < 4; ++t) {
            a[t]      = (__bf16)f0[t];
            a[4 + t]  = (__bf16)f1[t];
            a[8 + t]  = (__bf16)f2[t];
            a[12 + t] = (__bf16)f3[t];
        }
        #pragma unroll
        for (int c = 0; c < NCT; ++c) {
            v16bf b = *(const v16bf*)(Wp +
                ((long long)(c * ksteps + ks) * 32 + lane) * 16);
            acc[c] = __builtin_amdgcn_wmma_f32_16x16x32_bf16(
                false, a, false, b, (short)0, acc[c], false, false);
        }
    }

    // ---- guarded tail (clamped addresses + select; no exec branching) ----
    for (int ks = nFull; ks < ksteps; ++ks) {
        int k0 = ks * 32;
        v16bf a;
        #pragma unroll
        for (int t = 0; t < 16; ++t) {
            int kk = k0 + ((t < 8) ? (half * 8 + t) : (16 + half * 8 + (t - 8)));
            float x = Arow[kk < K ? kk : 0];
            a[t] = (__bf16)(kk < K ? x : 0.0f);
        }
        #pragma unroll
        for (int c = 0; c < NCT; ++c) {
            v16bf b = *(const v16bf*)(Wp +
                ((long long)(c * ksteps + ks) * 32 + lane) * 16);
            acc[c] = __builtin_amdgcn_wmma_f32_16x16x32_bf16(
                false, a, false, b, (short)0, acc[c], false, false);
        }
    }

    // ---- store D (f32 16x16: VGPR r -> row r + 8*half, col lane&15) ----
    #pragma unroll
    for (int c = 0; c < NCT; ++c) {
        int n = c * 16 + m;
        if (n < M) {
            #pragma unroll
            for (int r = 0; r < 8; ++r) {
                int orow = rt * 16 + r + half * 8;
                if (orow < N) Out[(long long)orow * ldo + n] = acc[c][r];
            }
        }
    }
}

// ---------------- aggregation -----------------------------------------------

// agg[i][h] = bias[h] + (1/deg[i]) * hw[i][h] for h < H; 0 for H <= h < STRIDE
__global__ void k_agg_init(float* __restrict__ agg, const float* __restrict__ hw,
                           const float* __restrict__ inv, const float* __restrict__ bias,
                           int N, int H) {
    long long idx = (long long)blockIdx.x * blockDim.x + threadIdx.x;
    if (idx < (long long)N * STRIDE) {
        int i = (int)(idx / STRIDE);
        int h = (int)(idx % STRIDE);
        float v = 0.0f;
        if (h < H) {
            float c = inv[i] * inv[i];
            v = bias[h] + c * hw[idx];
        }
        agg[idx] = v;
    }
}

// agg[dst] += c(e) * hw[src]; float4 gathers (rows are 128B aligned);
// edge list is streamed non-temporally, gathers/atomics stay in L2.
__global__ void k_agg_edges(float* __restrict__ agg, const float* __restrict__ hw,
                            const float* __restrict__ inv,
                            const long long* __restrict__ ei, long long E, int H) {
    long long e = (long long)blockIdx.x * blockDim.x + threadIdx.x;
    if (e < E) {
        int s = (int)__builtin_nontemporal_load(&ei[e]);
        int d = (int)__builtin_nontemporal_load(&ei[E + e]);
        float c = inv[s] * inv[d];
        const float* sp = hw + (long long)s * STRIDE;
        float* dp = agg + (long long)d * STRIDE;
        int h = 0;
        for (; h + 4 <= H; h += 4) {
            v4f v = *(const v4f*)(sp + h);
            atomic_add_f32(&dp[h + 0], c * v[0]);
            atomic_add_f32(&dp[h + 1], c * v[1]);
            atomic_add_f32(&dp[h + 2], c * v[2]);
            atomic_add_f32(&dp[h + 3], c * v[3]);
        }
        for (; h < H; ++h) atomic_add_f32(&dp[h], c * sp[h]);
    }
}

__global__ void k_relu(float* __restrict__ buf, long long n) {
    long long i = (long long)blockIdx.x * blockDim.x + threadIdx.x;
    if (i < n) buf[i] = fmaxf(buf[i], 0.0f);
}

__global__ void k_logsoftmax(float* __restrict__ out, const float* __restrict__ agg,
                             int N, int C) {
    int i = blockIdx.x * blockDim.x + threadIdx.x;
    if (i < N) {
        const float* r = agg + (long long)i * STRIDE;
        float mx = r[0];
        for (int c = 1; c < C; ++c) mx = fmaxf(mx, r[c]);
        float s = 0.0f;
        for (int c = 0; c < C; ++c) s += expf(r[c] - mx);
        float lse = mx + logf(s);
        for (int c = 0; c < C; ++c) out[(long long)i * C + c] = r[c] - lse;
    }
}

// ---------------- host side --------------------------------------------------

template <int NCT, bool NTA>
static void run_layer(const float* A, int lda, const float* W, int Kreal, int Kpad,
                      int H, const float* bias, const long long* ei, long long E,
                      const float* inv, __bf16* Wp, float* hw, float* agg,
                      int N, bool do_relu, hipStream_t stream) {
    const int t = 256;
    int ksteps = (Kreal + 31) / 32;              // Wp layout steps
    int nFull  = Kpad / 32;                      // unguarded steps
    int packTot = NCT * ksteps * 32 * 16;
    k_pack_w<<<(packTot + t - 1) / t, t, 0, stream>>>(Wp, W, Kreal, H, ksteps, NCT);

    int nRT = (N + 15) / 16;
    const int wavesPerBlock = 4;
    k_gemm_wmma<NCT, NTA><<<(nRT + wavesPerBlock - 1) / wavesPerBlock,
                            wavesPerBlock * 32, 0, stream>>>(
        A, lda, Wp, Kpad, nFull, ksteps, H, hw, STRIDE, N, nRT);

    long long ns = (long long)N * STRIDE;
    k_agg_init<<<(int)((ns + t - 1) / t), t, 0, stream>>>(agg, hw, inv, bias, N, H);
    k_agg_edges<<<(int)((E + t - 1) / t), t, 0, stream>>>(agg, hw, inv, ei, E, H);
    if (do_relu)
        k_relu<<<(int)((ns + t - 1) / t), t, 0, stream>>>(agg, ns);
}

extern "C" void kernel_launch(void* const* d_in, const int* in_sizes, int n_in,
                              void* d_out, int out_size, void* d_ws, size_t ws_size,
                              hipStream_t stream) {
    const float*     x  = (const float*)d_in[0];
    const long long* ei = (const long long*)d_in[1];   // int64 [2, E]
    const float*     W1 = (const float*)d_in[2];
    const float*     b1 = (const float*)d_in[3];
    const float*     W2 = (const float*)d_in[4];
    const float*     b2 = (const float*)d_in[5];
    const float*     W3 = (const float*)d_in[6];
    const float*     b3 = (const float*)d_in[7];
    float* out = (float*)d_out;

    const int H1 = in_sizes[3];                // 30
    const int H2 = in_sizes[5];                // 10
    const int C  = in_sizes[7];                // 7
    const int F  = in_sizes[2] / H1;           // 1433
    const int N  = in_sizes[0] / F;            // 100000
    const long long E = in_sizes[1] / 2;       // 3200000

    // Workspace: inv[N] | bufA[N*32] | bufB[N*32] | Wp (64B aligned, ~92KB)
    float* inv  = (float*)d_ws;
    float* bufA = inv + N;
    float* bufB = bufA + (size_t)N * STRIDE;
    uintptr_t wp = (uintptr_t)(bufB + (size_t)N * STRIDE);
    wp = (wp + 63) & ~(uintptr_t)63;
    __bf16* Wp = (__bf16*)wp;

    const int t = 256;
    k_init_deg<<<(N + t - 1) / t, t, 0, stream>>>(inv, N);
    k_accum_deg<<<(int)((E + t - 1) / t), t, 0, stream>>>(inv, ei, E);
    k_rsqrt<<<(N + t - 1) / t, t, 0, stream>>>(inv, N);

    // L1: A = x streamed non-temporally (K=1433 -> 44 full + 1 tail),
    //     2 col tiles fused per wave so x is read exactly once.
    run_layer<2, true>(x, F, W1, F, F, H1, b1, ei, E, inv,
                       Wp, bufA, bufB, N, true, stream);

    // L2: A = bufB (L2-resident, cols >= 30 zero) -> single unguarded K=32 step
    run_layer<1, false>(bufB, STRIDE, W2, H1, 32, H2, b2, ei, E, inv,
                        Wp, bufA, bufB, N, true, stream);

    // L3: A = bufB (L2-resident, cols >= 10 zero) -> single unguarded K=32 step
    run_layer<1, false>(bufB, STRIDE, W3, H2, 32, C, b3, ei, E, inv,
                        Wp, bufA, bufB, N, false, stream);

    k_logsoftmax<<<(N + t - 1) / t, t, 0, stream>>>(out, bufB, N, C);
}
